// DensityOnlyGATModuleNTV2_43516608643167
// MI455X (gfx1250) — matrix operand order repaired and checked by reading, not compiled, not run on hardware
//
#include <hip/hip_runtime.h>

typedef __attribute__((ext_vector_type(16))) __bf16 v16bf;
typedef __attribute__((ext_vector_type(8)))  float  v8f;

static constexpr int kN = 100000;
static constexpr int kE = 1600000;
static constexpr int kP = 2048;

#define DEV __device__ __forceinline__

DEV unsigned enc_f(float f) {
  unsigned u = __float_as_uint(f);
  return (u & 0x80000000u) ? ~u : (u | 0x80000000u);
}
DEV float dec_f(unsigned e) {
  return (e & 0x80000000u) ? __uint_as_float(e ^ 0x80000000u)
                           : __uint_as_float(~e);
}
DEV float lrelu(float x) { return x > 0.f ? x : 0.2f * x; }

// ---------------------------------------------------------------------------
// Weight prep: fp32 W[K,Nout] -> bf16 WT[Nout,Kp] (transposed, zero K-padded)
// ---------------------------------------------------------------------------
__global__ void k_prep_wt(const float* __restrict__ W, __bf16* __restrict__ WT,
                          int K, int Nout, int Kp) {
  int idx = blockIdx.x * blockDim.x + threadIdx.x;
  if (idx >= Nout * Kp) return;
  int n = idx / Kp, k = idx - n * Kp;
  WT[idx] = (__bf16)((k < K) ? W[(size_t)k * Nout + n] : 0.f);
}

// ce[h][k] = sum_c We[k, h*C+c] * ae[h,c]
__global__ void k_ce(const float* __restrict__ We, const float* __restrict__ ae,
                     float* __restrict__ ce, int H, int C) {
  int t = threadIdx.x;
  if (t >= H * 2) return;
  int h = t >> 1, kk = t & 1;
  float s = 0.f;
  for (int c = 0; c < C; ++c)
    s += We[(size_t)kk * H * C + h * C + c] * ae[h * C + c];
  ce[h * 2 + kk] = s;
}

// x0 = concat(node_features, tag, density[pm]) padded to 96, bf16
__global__ void k_build_x(const float* __restrict__ nf, const int* __restrict__ pm,
                          const float* __restrict__ pf, __bf16* __restrict__ xb) {
  int idx = blockIdx.x * blockDim.x + threadIdx.x;
  if (idx >= kN * 96) return;
  int n = idx / 96, k = idx - n * 96;
  float v;
  if (k < 64)       v = nf[(size_t)n * 64 + k];
  else if (k == 64) v = (n == kN - 1 || pm[n + 1] != pm[n]) ? 1.f : 0.f;
  else if (k == 65) v = pf[(size_t)pm[n] * 4];
  else              v = 0.f;
  xb[(size_t)n * 96 + k] = (__bf16)v;
}

// ---------------------------------------------------------------------------
// bf16 WMMA GEMM: Y[M,Nout] = X[M,Kp](bf16) @ WT[Nout,Kp](bf16, N-major)
// block = 256 thr = 8 waves; tile: 64 rows x Nout cols; K step 32.
// ---------------------------------------------------------------------------
template <int NT>
__global__ __launch_bounds__(256) void gemm_bf16_wmma(
    const __bf16* __restrict__ X, int ldx, const __bf16* __restrict__ WT,
    int ldw, float* __restrict__ Y, int M, int Kp, int Nout) {
  __shared__ __bf16 xs[64 * 32];
  __shared__ __bf16 wsm[256 * 32];
  const int tid = threadIdx.x;
  const int lane = tid & 31;
  const int wave = tid >> 5;
  const int msub = wave & 3;     // 4 x 16-row subtiles
  const int nhalf = wave >> 1 >> 1; // wave>>2: which N-half
  const int rowbase = blockIdx.x * 64;
  const int lgrp = lane >> 4;
  const int lrow = lane & 15;

  v8f acc[NT];
#pragma unroll
  for (int t = 0; t < NT; ++t) acc[t] = (v8f){0.f, 0.f, 0.f, 0.f, 0.f, 0.f, 0.f, 0.f};

  const unsigned* Xu = (const unsigned*)X;
  const unsigned* Wu = (const unsigned*)WT;
  unsigned* xsu = (unsigned*)xs;
  unsigned* wsu = (unsigned*)wsm;
  const int ldxu = ldx >> 1, ldwu = ldw >> 1;
  const int wTot = Nout * 16;  // uints in W tile

  for (int k0 = 0; k0 < Kp; k0 += 32) {
    __syncthreads();
#pragma unroll
    for (int i = 0; i < 4; ++i) {  // X tile: 1024 uints
      int idx = tid + i * 256;
      int r = idx >> 4, c = idx & 15;
      int row = rowbase + r;
      xsu[idx] = (row < M) ? Xu[(size_t)row * ldxu + (k0 >> 1) + c] : 0u;
    }
    for (int idx = tid; idx < wTot; idx += 256) {  // W tile: Nout*16 uints
      int r = idx >> 4, c = idx & 15;
      wsu[idx] = Wu[(size_t)r * ldwu + (k0 >> 1) + c];
    }
    __syncthreads();

    union { v16bf v; unsigned u[8]; } A;
#pragma unroll
    for (int j = 0; j < 8; ++j) {
      int ku = ((j < 4) ? j : j + 4) + lgrp * 4;  // A: K 0-7/16-23 | 8-15/24-31
      A.u[j] = xsu[(msub * 16 + lrow) * 16 + ku];
    }
#pragma unroll
    for (int t = 0; t < NT; ++t) {
      int n0 = nhalf * (Nout >> 1) + t * 16;
      union { v16bf v; unsigned u[8]; } B;
#pragma unroll
      for (int j = 0; j < 8; ++j) {
        int ku = ((j < 4) ? j : j + 4) + lgrp * 4;  // B: K 0-7/16-23 | 8-15/24-31
        B.u[j] = wsu[(n0 + lrow) * 16 + ku];
      }
      acc[t] = __builtin_amdgcn_wmma_f32_16x16x32_bf16(
          false, A.v, false, B.v, (short)0, acc[t], false, false);
    }
  }

#pragma unroll
  for (int t = 0; t < NT; ++t) {
    int n0 = nhalf * (Nout >> 1) + t * 16;
#pragma unroll
    for (int r = 0; r < 8; ++r) {
      int row = rowbase + msub * 16 + r + lgrp * 8;
      if (row < M) Y[(size_t)row * Nout + n0 + lrow] = acc[t][r];
    }
  }
}

// ---------------------------------------------------------------------------
// attention scalar passes
// ---------------------------------------------------------------------------
__global__ void k_alsald(const float* __restrict__ hb, const float* __restrict__ a_s,
                         const float* __restrict__ a_d, float* __restrict__ als,
                         float* __restrict__ ald, int H, int C, int HC) {
  int idx = blockIdx.x * blockDim.x + threadIdx.x;
  if (idx >= kN * H) return;
  int n = idx / H, hh = idx - n * H;
  const float* hp = hb + (size_t)n * HC + hh * C;
  const float* as_ = a_s + hh * C;
  const float* ad_ = a_d + hh * C;
  float s1 = 0.f, s2 = 0.f;
  for (int c = 0; c < C; ++c) {
    float v = hp[c];
    s1 += v * as_[c];
    s2 += v * ad_[c];
  }
  als[idx] = s1;
  ald[idx] = s2;
}

__global__ void k_edge_pre(const int* __restrict__ dst, const float* __restrict__ ea,
                           float* __restrict__ deg, float* __restrict__ lsum) {
  int e = blockIdx.x * blockDim.x + threadIdx.x;
  if (e >= kE) return;
  int d = dst[e];
  atomicAdd(deg + d, 1.f);
  atomicAdd(lsum + 2 * d, ea[2 * e]);
  atomicAdd(lsum + 2 * d + 1, ea[2 * e + 1]);
}

__global__ void k_node_init(const float* __restrict__ als, const float* __restrict__ ald,
                            const float* __restrict__ deg, const float* __restrict__ lsum,
                            const float* __restrict__ ce, float* __restrict__ slog,
                            unsigned* __restrict__ mx, int H) {
  int idx = blockIdx.x * blockDim.x + threadIdx.x;
  if (idx >= kN * H) return;
  int n = idx / H, hh = idx - n * H;
  float dg = fmaxf(deg[n], 1.f);
  float la0 = lsum[2 * n] / dg, la1 = lsum[2 * n + 1] / dg;
  float lg = lrelu(als[idx] + ald[idx] + la0 * ce[2 * hh] + la1 * ce[2 * hh + 1]);
  slog[idx] = lg;
  mx[idx] = enc_f(lg);  // self loop always present -> valid init for segment max
}

__global__ void k_edge_logit(const int* __restrict__ src, const int* __restrict__ dst,
                             const float* __restrict__ ea, const float* __restrict__ als,
                             const float* __restrict__ ald, const float* __restrict__ ce,
                             float* __restrict__ ev, unsigned* __restrict__ mx, int H) {
  int idx = blockIdx.x * blockDim.x + threadIdx.x;
  if (idx >= kE * H) return;
  int e = idx / H, hh = idx - e * H;
  int s = src[e], d = dst[e];
  float lg = lrelu(als[s * H + hh] + ald[d * H + hh] + ea[2 * e] * ce[2 * hh] +
                   ea[2 * e + 1] * ce[2 * hh + 1]);
  ev[idx] = lg;
  atomicMax(mx + d * H + hh, enc_f(lg));
}

__global__ void k_node_max(const unsigned* __restrict__ mx, const float* __restrict__ slog,
                           float* __restrict__ mf, float* __restrict__ denom, int H) {
  int idx = blockIdx.x * blockDim.x + threadIdx.x;
  if (idx >= kN * H) return;
  float m = dec_f(mx[idx]);
  mf[idx] = m;
  denom[idx] = __expf(slog[idx] - m);  // self-loop contribution initializes sum
}

__global__ void k_edge_exp(const int* __restrict__ dst, float* __restrict__ ev,
                           const float* __restrict__ mf, float* __restrict__ denom, int H) {
  int idx = blockIdx.x * blockDim.x + threadIdx.x;
  if (idx >= kE * H) return;
  int e = idx / H, hh = idx - e * H;
  int d = dst[e];
  float w = __expf(ev[idx] - mf[d * H + hh]);
  ev[idx] = w;
  atomicAdd(denom + d * H + hh, w);
}

// wave-per-edge gather h[src] * alpha -> scatter-add agg[dst]
template <int CPL>
__global__ __launch_bounds__(256) void k_edge_agg(
    const int* __restrict__ src, const int* __restrict__ dst,
    const float* __restrict__ ev, const float* __restrict__ denom,
    const float* __restrict__ hb, float* __restrict__ agg, int H, int HC) {
  int e = blockIdx.x * 8 + (threadIdx.x >> 5);
  if (e >= kE) return;
  int lane = threadIdx.x & 31;
  int s = src[e], d = dst[e];
  int c0 = lane * CPL;
  int hh = c0 >> 6;  // C == 64
  float den = denom[d * H + hh];
  den = den > 0.f ? den : 1.f;
  float alpha = ev[e * H + hh] / den;
  const float* hp = hb + (size_t)s * HC + c0;
  float* ap = agg + (size_t)d * HC + c0;
#pragma unroll
  for (int i = 0; i < CPL; ++i) atomicAdd(ap + i, hp[i] * alpha);
}

// wave-per-node: + self loop + bias, LayerNorm (+ReLU), emit bf16/fp32
template <int CPL>
__global__ __launch_bounds__(256) void k_node_final(
    const float* __restrict__ agg, const float* __restrict__ hb,
    const float* __restrict__ slog, const float* __restrict__ mf,
    const float* __restrict__ denom, const float* __restrict__ bias,
    const float* __restrict__ gam, const float* __restrict__ bet,
    __bf16* __restrict__ xb_out, int ldxb, float* __restrict__ xf_out,
    int relu, int H, int HC) {
  int n = blockIdx.x * 8 + (threadIdx.x >> 5);
  if (n >= kN) return;
  int lane = threadIdx.x & 31;
  int c0 = lane * CPL;
  int hh = c0 >> 6;
  float den = denom[n * H + hh];
  den = den > 0.f ? den : 1.f;
  float aself = __expf(slog[n * H + hh] - mf[n * H + hh]) / den;
  float v[CPL];
  float s = 0.f, s2 = 0.f;
#pragma unroll
  for (int i = 0; i < CPL; ++i) {
    int c = c0 + i;
    float x = agg[(size_t)n * HC + c] + hb[(size_t)n * HC + c] * aself + bias[c];
    v[i] = x;
    s += x;
    s2 += x * x;
  }
  for (int m = 16; m >= 1; m >>= 1) {
    s += __shfl_xor(s, m, 32);
    s2 += __shfl_xor(s2, m, 32);
  }
  float mu = s / (float)HC;
  float var = s2 / (float)HC - mu * mu;
  float r = rsqrtf(var + 1e-5f);
#pragma unroll
  for (int i = 0; i < CPL; ++i) {
    int c = c0 + i;
    float y = (v[i] - mu) * r * gam[c] + bet[c];
    if (relu) y = fmaxf(y, 0.f);
    if (xb_out) xb_out[(size_t)n * ldxb + c] = (__bf16)y;
    if (xf_out) xf_out[(size_t)n * HC + c] = y;
  }
}

// ---------------------------------------------------------------------------
// gate + pooling + final tiny GEMM
// ---------------------------------------------------------------------------
__global__ __launch_bounds__(256) void k_gate(const float* __restrict__ xf,
                                              const float* __restrict__ Wg,
                                              const float* __restrict__ bg,
                                              float* __restrict__ gate) {
  int n = blockIdx.x * 8 + (threadIdx.x >> 5);
  if (n >= kN) return;
  int lane = threadIdx.x & 31;
  float s = xf[(size_t)n * 64 + lane] * Wg[lane] +
            xf[(size_t)n * 64 + 32 + lane] * Wg[32 + lane];
  for (int m = 16; m >= 1; m >>= 1) s += __shfl_xor(s, m, 32);
  if (lane == 0) gate[n] = 1.f / (1.f + __expf(-(s + bg[0])));
}

DEV int node_cls(const int* pm, int n) {
  return (n == kN - 1 || pm[n + 1] != pm[n]) ? 1 : 0;  // 1 = solvent (tag==1)
}

__global__ void k_pool_max(const float* __restrict__ gate, const int* __restrict__ pm,
                           unsigned* __restrict__ mxp) {
  int n = blockIdx.x * blockDim.x + threadIdx.x;
  if (n >= kN) return;
  int cls = node_cls(pm, n);
  atomicMax(mxp + pm[n] * 2 + cls, enc_f(gate[n]));
}

__global__ void k_pool_sum(const float* __restrict__ gate, const int* __restrict__ pm,
                           const unsigned* __restrict__ mxp, float* __restrict__ sp) {
  int n = blockIdx.x * blockDim.x + threadIdx.x;
  if (n >= kN) return;
  int cls = node_cls(pm, n);
  int p = pm[n];
  float m = dec_f(mxp[p * 2 + cls]);
  atomicAdd(sp + p * 2 + cls, __expf(gate[n] - m));
}

__global__ __launch_bounds__(256) void k_pool_acc(
    const float* __restrict__ gate, const int* __restrict__ pm,
    const unsigned* __restrict__ mxp, const float* __restrict__ sp,
    const float* __restrict__ xf, float* __restrict__ pooled) {
  int n = blockIdx.x * 8 + (threadIdx.x >> 5);
  if (n >= kN) return;
  int lane = threadIdx.x & 31;
  int p = pm[n];
  int cls = node_cls(pm, n);
  float m = dec_f(mxp[p * 2 + cls]);
  float sv = sp[p * 2 + cls];
  sv = sv > 0.f ? sv : 1.f;
  float w = __expf(gate[n] - m) / sv;
  atomicAdd(pooled + (size_t)p * 128 + cls * 64 + lane,
            w * xf[(size_t)n * 64 + lane]);
  atomicAdd(pooled + (size_t)p * 128 + cls * 64 + 32 + lane,
            w * xf[(size_t)n * 64 + 32 + lane]);
}

__global__ void k_final(const float* __restrict__ pooled, const float* __restrict__ Wf,
                        const float* __restrict__ bf_, float* __restrict__ out) {
  int idx = blockIdx.x * blockDim.x + threadIdx.x;
  if (idx >= kP * 32) return;
  int p = idx >> 5, o = idx & 31;
  float s = bf_[o];
  for (int k = 0; k < 128; ++k) s += pooled[(size_t)p * 128 + k] * Wf[k * 32 + o];
  out[idx] = s;
}

// ---------------------------------------------------------------------------
extern "C" void kernel_launch(void* const* d_in, const int* in_sizes, int n_in,
                              void* d_out, int out_size, void* d_ws, size_t ws_size,
                              hipStream_t stream) {
  (void)in_sizes; (void)n_in; (void)out_size; (void)ws_size;
  const float* nf = (const float*)d_in[0];
  const int* ei = (const int*)d_in[1];
  const float* ea = (const float*)d_in[2];
  const int* pm = (const int*)d_in[3];
  const float* pf = (const float*)d_in[4];
  const float* Wmat[3] = {(const float*)d_in[5], (const float*)d_in[13], (const float*)d_in[21]};
  const float* ASp[3] = {(const float*)d_in[6], (const float*)d_in[14], (const float*)d_in[22]};
  const float* ADp[3] = {(const float*)d_in[7], (const float*)d_in[15], (const float*)d_in[23]};
  const float* WEp[3] = {(const float*)d_in[8], (const float*)d_in[16], (const float*)d_in[24]};
  const float* AEp[3] = {(const float*)d_in[9], (const float*)d_in[17], (const float*)d_in[25]};
  const float* Bp[3] = {(const float*)d_in[10], (const float*)d_in[18], (const float*)d_in[26]};
  const float* Gp[3] = {(const float*)d_in[11], (const float*)d_in[19], (const float*)d_in[27]};
  const float* BEp[3] = {(const float*)d_in[12], (const float*)d_in[20], (const float*)d_in[28]};
  const float* Wg = (const float*)d_in[29];
  const float* bg = (const float*)d_in[30];
  const float* Wf = (const float*)d_in[31];
  const float* bf_ = (const float*)d_in[32];
  float* out = (float*)d_out;

  const int* src = ei;
  const int* dstv = ei + kE;

  // --- workspace carve ---
  size_t off = 0;
  auto alloc = [&](size_t bytes) -> void* {
    void* p = (char*)d_ws + off;
    off = (off + bytes + 255) & ~(size_t)255;
    return p;
  };
  __bf16* WT1 = (__bf16*)alloc((size_t)256 * 96 * 2);
  __bf16* WT2 = (__bf16*)alloc((size_t)256 * 256 * 2);
  __bf16* WT3 = (__bf16*)alloc((size_t)64 * 256 * 2);
  float* ceb[3] = {(float*)alloc(64), (float*)alloc(64), (float*)alloc(64)};
  __bf16* xb = (__bf16*)alloc((size_t)kN * 256 * 2);
  float* hbuf = (float*)alloc((size_t)kN * 256 * 4);
  float* agg = (float*)alloc((size_t)kN * 256 * 4);
  float* als = (float*)alloc((size_t)kN * 4 * 4);
  float* ald = (float*)alloc((size_t)kN * 4 * 4);
  float* slog = (float*)alloc((size_t)kN * 4 * 4);
  float* mf = (float*)alloc((size_t)kN * 4 * 4);
  float* denom = (float*)alloc((size_t)kN * 4 * 4);
  unsigned* mx = (unsigned*)alloc((size_t)kN * 4 * 4);
  float* deg = (float*)alloc((size_t)kN * 4);
  float* lsum = (float*)alloc((size_t)kN * 2 * 4);
  float* ev = (float*)alloc((size_t)kE * 4 * 4);
  float* gate = (float*)alloc((size_t)kN * 4);
  unsigned* mxp = (unsigned*)alloc((size_t)kP * 2 * 4);
  float* sp = (float*)alloc((size_t)kP * 2 * 4);
  float* pooled = (float*)alloc((size_t)kP * 128 * 4);

  auto cdiv = [](long a, long b) { return (int)((a + b - 1) / b); };

  // --- one-time prep ---
  k_prep_wt<<<cdiv(256 * 96, 256), 256, 0, stream>>>(Wmat[0], WT1, 66, 256, 96);
  k_prep_wt<<<cdiv(256 * 256, 256), 256, 0, stream>>>(Wmat[1], WT2, 256, 256, 256);
  k_prep_wt<<<cdiv(64 * 256, 256), 256, 0, stream>>>(Wmat[2], WT3, 256, 64, 256);
  k_ce<<<1, 32, 0, stream>>>(WEp[0], AEp[0], ceb[0], 4, 64);
  k_ce<<<1, 32, 0, stream>>>(WEp[1], AEp[1], ceb[1], 4, 64);
  k_ce<<<1, 32, 0, stream>>>(WEp[2], AEp[2], ceb[2], 1, 64);
  k_build_x<<<cdiv((long)kN * 96, 256), 256, 0, stream>>>(nf, pm, pf, xb);

  const __bf16* WTl[3] = {WT1, WT2, WT3};
  const int Kpl[3] = {96, 256, 256};
  const int Hl[3] = {4, 4, 1};

  for (int L = 0; L < 3; ++L) {
    const int H = Hl[L];
    const int HC = H * 64;
    const int Nout = HC;
    const int Kp = Kpl[L];
    const int ldx = (L == 0) ? 96 : 256;

    // GEMM: hbuf = xb @ W  (bf16 WMMA, fp32 accumulate)
    if (L < 2)
      gemm_bf16_wmma<8><<<cdiv(kN, 64), 256, 0, stream>>>(xb, ldx, WTl[L], Kp,
                                                          hbuf, kN, Kp, Nout);
    else
      gemm_bf16_wmma<2><<<cdiv(kN, 64), 256, 0, stream>>>(xb, ldx, WTl[L], Kp,
                                                          hbuf, kN, Kp, Nout);

    k_alsald<<<cdiv((long)kN * H, 256), 256, 0, stream>>>(hbuf, ASp[L], ADp[L],
                                                          als, ald, H, 64, HC);
    hipMemsetAsync(deg, 0, (size_t)kN * 4, stream);
    hipMemsetAsync(lsum, 0, (size_t)kN * 8, stream);
    k_edge_pre<<<cdiv(kE, 256), 256, 0, stream>>>(dstv, ea, deg, lsum);
    k_node_init<<<cdiv((long)kN * H, 256), 256, 0, stream>>>(als, ald, deg, lsum,
                                                             ceb[L], slog, mx, H);
    k_edge_logit<<<cdiv((long)kE * H, 256), 256, 0, stream>>>(src, dstv, ea, als,
                                                              ald, ceb[L], ev, mx, H);
    k_node_max<<<cdiv((long)kN * H, 256), 256, 0, stream>>>(mx, slog, mf, denom, H);
    k_edge_exp<<<cdiv((long)kE * H, 256), 256, 0, stream>>>(dstv, ev, mf, denom, H);
    hipMemsetAsync(agg, 0, (size_t)kN * HC * 4, stream);
    if (L < 2) {
      k_edge_agg<8><<<cdiv(kE, 8), 256, 0, stream>>>(src, dstv, ev, denom, hbuf,
                                                     agg, H, HC);
      k_node_final<8><<<cdiv(kN, 8), 256, 0, stream>>>(
          agg, hbuf, slog, mf, denom, Bp[L], Gp[L], BEp[L],
          /*xb_out=*/xb, /*ldxb=*/256, /*xf_out=*/nullptr, /*relu=*/1, H, HC);
    } else {
      k_edge_agg<2><<<cdiv(kE, 8), 256, 0, stream>>>(src, dstv, ev, denom, hbuf,
                                                     agg, H, HC);
      // layer 3: no relu; fp32 result written in-place into hbuf for pooling
      k_node_final<2><<<cdiv(kN, 8), 256, 0, stream>>>(
          agg, hbuf, slog, mf, denom, Bp[L], Gp[L], BEp[L],
          /*xb_out=*/nullptr, /*ldxb=*/0, /*xf_out=*/hbuf, /*relu=*/0, H, HC);
    }
  }

  // gate + masked softmax pools
  k_gate<<<cdiv(kN, 8), 256, 0, stream>>>(hbuf, Wg, bg, gate);
  hipMemsetAsync(mxp, 0, (size_t)kP * 2 * 4, stream);  // enc(-inf) ~ 0
  hipMemsetAsync(sp, 0, (size_t)kP * 2 * 4, stream);
  hipMemsetAsync(pooled, 0, (size_t)kP * 128 * 4, stream);
  k_pool_max<<<cdiv(kN, 256), 256, 0, stream>>>(gate, pm, mxp);
  k_pool_sum<<<cdiv(kN, 256), 256, 0, stream>>>(gate, pm, mxp, sp);
  k_pool_acc<<<cdiv(kN, 8), 256, 0, stream>>>(gate, pm, mxp, sp, hbuf, pooled);
  k_final<<<cdiv((long)kP * 32, 256), 256, 0, stream>>>(pooled, Wf, bf_, out);
}
